// DifferentialMultiHeadSelfAttention_83021717832618
// MI455X (gfx1250) — compile-verified, matrix operand order checked by reading, not run on hardware
//
#include <hip/hip_runtime.h>
#include <hip/hip_bf16.h>
#include <math.h>

// Problem dims (compile-time)
#define B_    2
#define S_    2048
#define E_    512
#define H_    8
#define D_    512
#define HALF_ 256
#define HD_   (H_ * D_)

typedef __attribute__((ext_vector_type(16))) __bf16 v16bf;
typedef __attribute__((ext_vector_type(8)))  __bf16 v8bf;
typedef __attribute__((ext_vector_type(8)))  float  v8f;

// ---- helpers -------------------------------------------------------------

static __device__ __forceinline__ __bf16 f2bf(float f) {
  union { float f; unsigned u; } c; c.f = f;
  unsigned r = c.u + 0x7FFFu + ((c.u >> 16) & 1u);   // round-to-nearest-even
  union { unsigned short s; __bf16 b; } o; o.s = (unsigned short)(r >> 16);
  return o.b;
}

// Async global->LDS copy of 16B (one b128 transfer per lane), ASYNCcnt-tracked.
static __device__ __forceinline__ void async_b128(const __bf16* g, __bf16* l) {
  unsigned lds = (unsigned)(uintptr_t)l;       // generic shared addr[31:0] == LDS offset
  asm volatile("global_load_async_to_lds_b128 %0, %1, off"
               :: "v"(lds), "v"(g) : "memory");
}
static __device__ __forceinline__ void wait_async0() {
  asm volatile("s_wait_asynccnt 0" ::: "memory");
}
static __device__ __forceinline__ void wait_ds0() {
  asm volatile("s_wait_dscnt 0" ::: "memory");
}

// Stage a 16-row x 512-col bf16 panel (16 KB) global -> LDS, 4 b128 per thread
// (expects blockDim.x == 256).
static __device__ __forceinline__ void stage_panelA(const __bf16* __restrict__ g,
                                                    int gstride, __bf16* l) {
  const int t = threadIdx.x;
  const int col  = (t * 8) & 511;
  const int row0 = (t * 8) >> 9;
#pragma unroll
  for (int i = 0; i < 4; ++i) {
    const int row = row0 + i * 4;
    async_b128(g + (size_t)row * gstride + col, l + row * 512 + col);
  }
}

// Load one 16x32 bf16 WMMA operand tile from a row-major [rows, stride] array.
// A-operand layout (and the symmetric B layout on a [N,K] array):
//   lane l -> row = row0 + (l & 15); chunks at K = col0 + (l>>4)*8 and +16.
static __device__ __forceinline__ v16bf load_tile_g(const __bf16* __restrict__ base,
                                                    int row0, int stride, int col0) {
  const int lane = threadIdx.x & 31;
  const int rr = lane & 15, sel = lane >> 4;
  const __bf16* p = base + (size_t)(row0 + rr) * stride + col0 + sel * 8;
  v8bf lo = *(const v8bf*)p;
  v8bf hi = *(const v8bf*)(p + 16);
  v16bf o;
#pragma unroll
  for (int i = 0; i < 8; ++i) { o[i] = lo[i]; o[i + 8] = hi[i]; }
  return o;
}

// A-operand tile from a 16x512 LDS panel.
static __device__ __forceinline__ v16bf load_tile_lds(const __bf16* base, int col0) {
  const int lane = threadIdx.x & 31;
  const int rr = lane & 15, sel = lane >> 4;
  const __bf16* p = base + rr * 512 + col0 + sel * 8;
  v8bf lo = *(const v8bf*)p;
  v8bf hi = *(const v8bf*)(p + 16);
  v16bf o;
#pragma unroll
  for (int i = 0; i < 8; ++i) { o[i] = lo[i]; o[i + 8] = hi[i]; }
  return o;
}

static __device__ __forceinline__ v8f wmma_bf16(v16bf a, v16bf b, v8f c) {
  // D(f32 16x16) = A(16x32 bf16) x B(32x16 bf16) + C
  return __builtin_amdgcn_wmma_f32_16x16x32_bf16(false, a, false, b, (short)0, c,
                                                 false, false);
}

// ---- prep kernels --------------------------------------------------------

__global__ void cvt_bf16_kernel(const float* __restrict__ in, __bf16* __restrict__ out, int n) {
  int i = blockIdx.x * blockDim.x + threadIdx.x;
  if (i < n) out[i] = f2bf(in[i]);
}

// in[hb][r][c] (f32) -> out[hb][c][r] (bf16)
__global__ void transpose_bf16_kernel(const float* __restrict__ in, __bf16* __restrict__ out,
                                      int nb, int R, int C) {
  int i = blockIdx.x * blockDim.x + threadIdx.x;
  int plane = R * C;
  if (i >= nb * plane) return;
  int hb = i / plane;
  int rem = i - hb * plane;
  int r = rem / C, c = rem - r * C;
  out[(size_t)hb * plane + (size_t)c * R + r] = f2bf(in[i]);
}

__global__ void lam_kernel(const float* __restrict__ lq1, const float* __restrict__ lk1,
                           const float* __restrict__ lq2, const float* __restrict__ lk2,
                           const float* __restrict__ lam_init_p, float* __restrict__ lam) {
  int h = threadIdx.x;
  if (h >= H_) return;
  float s1 = 0.f, s2 = 0.f;
  for (int d = 0; d < D_; ++d) {
    s1 += lq1[h * D_ + d] * lk1[h * D_ + d];
    s2 += lq2[h * D_ + d] * lk2[h * D_ + d];
  }
  lam[h] = __expf(s1) - __expf(s2) + lam_init_p[h];
}

// ---- QKV projection (WMMA GEMM) ------------------------------------------
// q,k: bf16 [B*H][S][D] row-major;  v: bf16 transposed [B*H][D][S]
// A panel (16 x E) staged once into LDS via async load and shared by 8 waves.
__global__ __launch_bounds__(256) void qkv_kernel(
    const __bf16* __restrict__ xb,
    const __bf16* __restrict__ WqT, const __bf16* __restrict__ WkT, const __bf16* __restrict__ WvT,
    const float* __restrict__ bq, const float* __restrict__ bk, const float* __restrict__ bv,
    __bf16* __restrict__ qO, __bf16* __restrict__ kO, __bf16* __restrict__ vTO) {
  __shared__ __attribute__((aligned(16))) __bf16 aLds[16 * E_];   // 16 KB

  const int sel  = blockIdx.z;          // 0:q 1:k 2:v
  const int h    = blockIdx.y;
  const int m0   = blockIdx.x * 16;     // row tile over B*S
  const int wave = threadIdx.x >> 5;
  const int lane = threadIdx.x & 31;
  const int d0   = wave * 64;           // 8 waves x 64 cols = D

  const __bf16* WT   = (sel == 0 ? WqT : sel == 1 ? WkT : WvT) + (size_t)h * D_ * E_;
  const float*  bias = (sel == 0 ? bq : sel == 1 ? bk : bv) + h * D_;

  stage_panelA(xb + (size_t)m0 * E_, E_, aLds);
  wait_async0();
  __syncthreads();

  v8f c0 = {}, c1 = {}, c2 = {}, c3 = {};
  // double-buffered B tiles
  v16bf b0 = load_tile_g(WT, d0 +  0, E_, 0);
  v16bf b1 = load_tile_g(WT, d0 + 16, E_, 0);
  v16bf b2 = load_tile_g(WT, d0 + 32, E_, 0);
  v16bf b3 = load_tile_g(WT, d0 + 48, E_, 0);
#pragma unroll 4
  for (int kc = 0; kc < E_ / 32; ++kc) {
    const int kn = (kc + 1 < E_ / 32) ? (kc + 1) * 32 : kc * 32;
    v16bf n0 = load_tile_g(WT, d0 +  0, E_, kn);
    v16bf n1 = load_tile_g(WT, d0 + 16, E_, kn);
    v16bf n2 = load_tile_g(WT, d0 + 32, E_, kn);
    v16bf n3 = load_tile_g(WT, d0 + 48, E_, kn);
    v16bf a = load_tile_lds(aLds, kc * 32);
    c0 = wmma_bf16(a, b0, c0);
    c1 = wmma_bf16(a, b1, c1);
    c2 = wmma_bf16(a, b2, c2);
    c3 = wmma_bf16(a, b3, c3);
    b0 = n0; b1 = n1; b2 = n2; b3 = n3;
  }

  const int lh = lane >> 4, ln = lane & 15;
  v8f cc[4] = {c0, c1, c2, c3};
#pragma unroll
  for (int j = 0; j < 4; ++j) {
    int d = d0 + j * 16 + ln;
    float bvv = bias[d];
#pragma unroll
    for (int r = 0; r < 8; ++r) {
      int bs = m0 + r + 8 * lh;
      int b  = bs >> 11;                 // /S_
      int s  = bs & (S_ - 1);
      size_t bh = (size_t)(b * H_ + h);
      float val = cc[j][r] + bvv;
      if (sel == 2)      vTO[(bh * D_ + d) * S_ + s] = f2bf(val);
      else if (sel == 0) qO[(bh * S_ + s) * D_ + d]  = f2bf(val);
      else               kO[(bh * S_ + s) * D_ + d]  = f2bf(val);
    }
  }
}

// ---- fused two-pass differential flash attention -------------------------
// One wave per block: 16 query rows, full D=512 accumulator (256 VGPRs).
// pass0: softmax(q1 k1^T) V -> LDS stash; pass1: softmax(q2 k2^T) V,
// combine o1 - lam*o2, RMS norm over D, scale, write bf16 [B*S][H*D].
__global__ __launch_bounds__(32)
__attribute__((amdgpu_waves_per_eu(1)))
void attn_kernel(
    const __bf16* __restrict__ q, const __bf16* __restrict__ k, const __bf16* __restrict__ vT,
    const float* __restrict__ lam, const float* __restrict__ rms_w,
    __bf16* __restrict__ ob) {
  __shared__ __attribute__((aligned(16))) float  stash[16 * D_];   // 32 KB
  __shared__ __attribute__((aligned(16))) __bf16 pbuf[16 * 32];    // 1 KB

  const int bh  = blockIdx.y;
  const int h   = bh & (H_ - 1);
  const int b   = bh >> 3;
  const int sm0 = blockIdx.x * 16;
  const int lane = threadIdx.x & 31;
  const int lh = lane >> 4, ln = lane & 15;
  const float scale = 0.0625f;                 // 1/sqrt(256)
  const size_t qkBase = (size_t)bh * S_ * D_;
  const size_t vBase  = (size_t)bh * D_ * S_;
  const int nblocks = sm0 / 32 + 1;            // causal: key blocks of 32

  for (int pass = 0; pass < 2; ++pass) {
    const int coff = pass * HALF_;

    v16bf aQ[8];
#pragma unroll
    for (int kc = 0; kc < 8; ++kc)
      aQ[kc] = load_tile_g(q + qkBase, sm0, D_, coff + kc * 32);

    v8f acc[32];
#pragma unroll
    for (int nt = 0; nt < 32; ++nt) { v8f z = {}; acc[nt] = z; }
    float mrow[8], lrow[8];
#pragma unroll
    for (int r = 0; r < 8; ++r) { mrow[r] = -3.0e38f; lrow[r] = 0.f; }

    for (int tb = 0; tb < nblocks; ++tb) {
      const int t0 = tb * 32;
      // prefetch next key block (one byte per row -> whole cachelines)
      if (tb + 1 < nblocks)
        __builtin_prefetch(k + qkBase + (size_t)(t0 + 32 + lane) * D_ + coff, 0, 1);

      // ---- scores: 16x32 tile of q_half . k_half^T  (double-buffered K)
      v8f s0 = {}, s1 = {};
      v16bf kb0 = load_tile_g(k + qkBase, t0,      D_, coff);
      v16bf kb1 = load_tile_g(k + qkBase, t0 + 16, D_, coff);
#pragma unroll
      for (int kc = 0; kc < 8; ++kc) {
        const int kn = coff + ((kc + 1 < 8) ? (kc + 1) : kc) * 32;
        v16bf n0 = load_tile_g(k + qkBase, t0,      D_, kn);
        v16bf n1 = load_tile_g(k + qkBase, t0 + 16, D_, kn);
        s0 = wmma_bf16(aQ[kc], kb0, s0);
        s1 = wmma_bf16(aQ[kc], kb1, s1);
        kb0 = n0; kb1 = n1;
      }
      // scale + causal mask
#pragma unroll
      for (int r = 0; r < 8; ++r) {
        int row = sm0 + r + 8 * lh;
        s0[r] = ((t0 + ln)      > row) ? -3.0e38f : s0[r] * scale;
        s1[r] = ((t0 + 16 + ln) > row) ? -3.0e38f : s1[r] * scale;
      }
      // online softmax update (row reductions across the 16-lane half)
      float alpha[8];
#pragma unroll
      for (int r = 0; r < 8; ++r) {
        float mx = fmaxf(s0[r], s1[r]);
#pragma unroll
        for (int msk = 8; msk >= 1; msk >>= 1) mx = fmaxf(mx, __shfl_xor(mx, msk, 32));
        float mnew = fmaxf(mrow[r], mx);
        alpha[r] = __expf(mrow[r] - mnew);
        mrow[r] = mnew;
        s0[r] = __expf(s0[r] - mnew);
        s1[r] = __expf(s1[r] - mnew);
        float rs = s0[r] + s1[r];
#pragma unroll
        for (int msk = 8; msk >= 1; msk >>= 1) rs += __shfl_xor(rs, msk, 32);
        lrow[r] = lrow[r] * alpha[r] + rs;
      }
#pragma unroll
      for (int nt = 0; nt < 32; ++nt)
#pragma unroll
        for (int r = 0; r < 8; ++r) acc[nt][r] *= alpha[r];

      // stage P (C layout) to LDS, reload in A-operand layout
#pragma unroll
      for (int r = 0; r < 8; ++r) {
        int row = r + 8 * lh;
        pbuf[row * 32 + ln]      = f2bf(s0[r]);
        pbuf[row * 32 + 16 + ln] = f2bf(s1[r]);
      }
      wait_ds0();
      v16bf pA;
      {
        const __bf16* pp = &pbuf[ln * 32 + lh * 8];
        v8bf lo = *(const v8bf*)pp;
        v8bf hi = *(const v8bf*)(pp + 16);
#pragma unroll
        for (int i = 0; i < 8; ++i) { pA[i] = lo[i]; pA[i + 8] = hi[i]; }
      }
      // ---- acc(16x512) += P(16x32) @ V(32x512), V from transposed [D][S]
      v16bf bV = load_tile_g(vT + vBase, 0, S_, t0);
#pragma unroll
      for (int nt = 0; nt < 32; ++nt) {
        v16bf bVn = load_tile_g(vT + vBase, ((nt + 1 < 32) ? nt + 1 : nt) * 16, S_, t0);
        acc[nt] = wmma_bf16(pA, bV, acc[nt]);
        bV = bVn;
      }
    }

    // normalize by softmax denominator
    float linv[8];
#pragma unroll
    for (int r = 0; r < 8; ++r) linv[r] = 1.0f / lrow[r];
#pragma unroll
    for (int nt = 0; nt < 32; ++nt)
#pragma unroll
      for (int r = 0; r < 8; ++r) acc[nt][r] *= linv[r];

    if (pass == 0) {
#pragma unroll
      for (int nt = 0; nt < 32; ++nt)
#pragma unroll
        for (int r = 0; r < 8; ++r)
          stash[(r + 8 * lh) * D_ + nt * 16 + ln] = acc[nt][r];
      wait_ds0();
    } else {
      const float lm = lam[h];
      float ss[8];
#pragma unroll
      for (int r = 0; r < 8; ++r) ss[r] = 0.f;
#pragma unroll
      for (int nt = 0; nt < 32; ++nt)
#pragma unroll
        for (int r = 0; r < 8; ++r) {
          int idx = (r + 8 * lh) * D_ + nt * 16 + ln;
          float o = stash[idx] - lm * acc[nt][r];     // o1 - lam*o2
          stash[idx] = o;
          ss[r] += o * o;
        }
      wait_ds0();
      float rn[8];
#pragma unroll
      for (int r = 0; r < 8; ++r) {
        float t = ss[r];
#pragma unroll
        for (int msk = 8; msk >= 1; msk >>= 1) t += __shfl_xor(t, msk, 32);
        rn[r] = rsqrtf(t * (1.0f / (float)D_) + 1.1920929e-7f);
      }
#pragma unroll
      for (int nt = 0; nt < 32; ++nt)
#pragma unroll
        for (int r = 0; r < 8; ++r) {
          int d   = nt * 16 + ln;
          int row = r + 8 * lh;
          float o = stash[row * D_ + d] * rn[r] * 0.2f * rms_w[h * D_ + d];
          int s = sm0 + row;
          ob[((size_t)(b * S_ + s)) * HD_ + h * D_ + d] = f2bf(o);
        }
    }
  }
}

// ---- output projection: out[B*S][E] = ob @ Wo + bo -----------------------
// A (16 x 4096) streamed through a double-buffered async LDS pipeline in
// 16x512 chunks; each chunk overlaps the WMMA of the previous one.
__global__ __launch_bounds__(256) void out_gemm_kernel(
    const __bf16* __restrict__ obv, const __bf16* __restrict__ WoT,
    const float* __restrict__ bo, float* __restrict__ out) {
  __shared__ __attribute__((aligned(16))) __bf16 aLds[2][16 * 512];   // 32 KB

  const int m0   = blockIdx.x * 16;
  const int wave = threadIdx.x >> 5;
  const int lane = threadIdx.x & 31;
  const int e0   = wave * 64;            // 8 waves x 64 = E

  v8f c0 = {}, c1 = {}, c2 = {}, c3 = {};
  stage_panelA(obv + (size_t)m0 * HD_, HD_, aLds[0]);

  for (int ck = 0; ck < 8; ++ck) {
    wait_async0();
    __syncthreads();
    if (ck + 1 < 8)
      stage_panelA(obv + (size_t)m0 * HD_ + (ck + 1) * 512, HD_, aLds[(ck + 1) & 1]);

    const __bf16* aP = aLds[ck & 1];
    const int kbase = ck * 512;
    v16bf b0 = load_tile_g(WoT, e0 +  0, HD_, kbase);
    v16bf b1 = load_tile_g(WoT, e0 + 16, HD_, kbase);
    v16bf b2 = load_tile_g(WoT, e0 + 32, HD_, kbase);
    v16bf b3 = load_tile_g(WoT, e0 + 48, HD_, kbase);
#pragma unroll 4
    for (int kc = 0; kc < 16; ++kc) {
      const int kn = kbase + ((kc + 1 < 16) ? (kc + 1) : kc) * 32;
      v16bf n0 = load_tile_g(WoT, e0 +  0, HD_, kn);
      v16bf n1 = load_tile_g(WoT, e0 + 16, HD_, kn);
      v16bf n2 = load_tile_g(WoT, e0 + 32, HD_, kn);
      v16bf n3 = load_tile_g(WoT, e0 + 48, HD_, kn);
      v16bf a = load_tile_lds(aP, kc * 32);
      c0 = wmma_bf16(a, b0, c0);
      c1 = wmma_bf16(a, b1, c1);
      c2 = wmma_bf16(a, b2, c2);
      c3 = wmma_bf16(a, b3, c3);
      b0 = n0; b1 = n1; b2 = n2; b3 = n3;
    }
  }

  const int lh = lane >> 4, ln = lane & 15;
  v8f cc[4] = {c0, c1, c2, c3};
#pragma unroll
  for (int j = 0; j < 4; ++j) {
    int e = e0 + j * 16 + ln;
    float bov = bo[e];
#pragma unroll
    for (int r = 0; r < 8; ++r) {
      int bs = m0 + r + 8 * lh;
      out[(size_t)bs * E_ + e] = cc[j][r] + bov;
    }
  }
}

// ---- launcher ------------------------------------------------------------

extern "C" void kernel_launch(void* const* d_in, const int* in_sizes, int n_in,
                              void* d_out, int out_size, void* d_ws, size_t ws_size,
                              hipStream_t stream) {
  const float* x    = (const float*)d_in[0];
  /* d_in[1] = causal mask (bool), implied by kernel */
  const float* Wq   = (const float*)d_in[2];
  const float* bq   = (const float*)d_in[3];
  const float* Wk   = (const float*)d_in[4];
  const float* bk   = (const float*)d_in[5];
  const float* Wv   = (const float*)d_in[6];
  const float* bv   = (const float*)d_in[7];
  const float* lq1  = (const float*)d_in[8];
  const float* lk1  = (const float*)d_in[9];
  const float* lq2  = (const float*)d_in[10];
  const float* lk2  = (const float*)d_in[11];
  const float* lamp = (const float*)d_in[12];
  const float* rmsw = (const float*)d_in[13];
  const float* Wo   = (const float*)d_in[14];
  const float* bo   = (const float*)d_in[15];
  float* out = (float*)d_out;

  char* ws = (char*)d_ws;
  size_t off = 0;
  auto alloc = [&](size_t bytes) -> void* {
    void* p = (void*)(ws + off);
    off += (bytes + 255) & ~(size_t)255;
    return p;
  };
  __bf16* xb   = (__bf16*)alloc((size_t)B_ * S_ * E_ * 2);
  __bf16* WqT  = (__bf16*)alloc((size_t)H_ * D_ * E_ * 2);
  __bf16* WkT  = (__bf16*)alloc((size_t)H_ * D_ * E_ * 2);
  __bf16* WvT  = (__bf16*)alloc((size_t)H_ * D_ * E_ * 2);
  __bf16* WoT  = (__bf16*)alloc((size_t)E_ * HD_ * 2);
  __bf16* qB   = (__bf16*)alloc((size_t)B_ * H_ * S_ * D_ * 2);
  __bf16* kB   = (__bf16*)alloc((size_t)B_ * H_ * S_ * D_ * 2);
  __bf16* vTB  = (__bf16*)alloc((size_t)B_ * H_ * D_ * S_ * 2);
  __bf16* obB  = (__bf16*)alloc((size_t)B_ * S_ * HD_ * 2);
  float*  lamB = (float*)alloc(H_ * sizeof(float));

  int n;
  n = B_ * S_ * E_;
  cvt_bf16_kernel<<<(n + 255) / 256, 256, 0, stream>>>(x, xb, n);
  n = H_ * E_ * D_;
  transpose_bf16_kernel<<<(n + 255) / 256, 256, 0, stream>>>(Wq, WqT, H_, E_, D_);
  transpose_bf16_kernel<<<(n + 255) / 256, 256, 0, stream>>>(Wk, WkT, H_, E_, D_);
  transpose_bf16_kernel<<<(n + 255) / 256, 256, 0, stream>>>(Wv, WvT, H_, E_, D_);
  n = HD_ * E_;
  transpose_bf16_kernel<<<(n + 255) / 256, 256, 0, stream>>>(Wo, WoT, 1, HD_, E_);
  lam_kernel<<<1, 32, 0, stream>>>(lq1, lk1, lq2, lk2, lamp, lamB);

  qkv_kernel<<<dim3(B_ * S_ / 16, H_, 3), 256, 0, stream>>>(
      xb, WqT, WkT, WvT, bq, bk, bv, qB, kB, vTB);

  attn_kernel<<<dim3(S_ / 16, B_ * H_), 32, 0, stream>>>(
      qB, kB, vTB, lamB, rmsw, obB);

  out_gemm_kernel<<<dim3(B_ * S_ / 16), 256, 0, stream>>>(obB, WoT, bo, out);
}
